// Attention_61332132986998
// MI455X (gfx1250) — matrix operand order repaired and checked by reading, not compiled, not run on hardware
//
#include <hip/hip_runtime.h>
#include <hip/hip_bf16.h>

// MI455X (gfx1250) attention for Swin-style windows: B=4096, N=49, C=256, 8 heads x 32.
// All GEMMs via v_wmma_f32_16x16x32_f16 (f16 in, f32 accum). Softmax / rel-pos MLP in f32.
// Weight GEMMs are N-chunked (not M-chunked) so each block streams each weight byte
// exactly once from L2: 512 KB/block -> ~2.1 GB/call, on par with the 410 MB HBM floor.
// Workspace layout (needs >= 640 KB):
//   [0, 384K)        : QKV weights f16, B-fragment swizzled  (48 ntiles x 8 ksteps x 32 lanes x 16 halfs)
//   [384K, 512K)     : proj weights f16, B-fragment swizzled (16 ntiles x 8 ksteps x 32 lanes x 16 halfs)
//   [512K, 640K)     : padded rel-pos bias f32 [8][64][64]

typedef _Float16 h16v __attribute__((ext_vector_type(16)));
typedef _Float16 h8v  __attribute__((ext_vector_type(8)));
typedef _Float16 h4v  __attribute__((ext_vector_type(4)));
typedef float    f8v  __attribute__((ext_vector_type(8)));
typedef float    f4v  __attribute__((ext_vector_type(4)));

__device__ __forceinline__ f8v wmma16(h16v a, h16v b, f8v c) {
  return __builtin_amdgcn_wmma_f32_16x16x32_f16(false, a, false, b, (short)0, c, false, false);
}

// A fragment (16x32 f16) from row-major LDS: lane<16 gets K {k0..k0+15}, lane>=16 gets K {k0+16..k0+31}.
__device__ __forceinline__ h16v ldsA(const _Float16* base, int row0, int stride, int k0, int lane) {
  const int m  = row0 + (lane & 15);
  const int hi = (lane >> 4) & 1;
  const _Float16* p = base + m * stride + k0 + hi * 16;
  h8v lo = *(const h8v*)p;
  h8v hp = *(const h8v*)(p + 8);
  h16v r;
#pragma unroll
  for (int i = 0; i < 8; ++i) { r[i] = lo[i]; r[i + 8] = hp[i]; }
  return r;
}

// B fragment: 16 contiguous halfs per lane (pointer already lane-resolved).
__device__ __forceinline__ h16v ldB32(const _Float16* p) {
  h8v lo = *(const h8v*)p;
  h8v hp = *(const h8v*)(p + 8);
  h16v r;
#pragma unroll
  for (int i = 0; i < 8; ++i) { r[i] = lo[i]; r[i + 8] = hp[i]; }
  return r;
}

// ---------------- weight convert + B-fragment swizzle ----------------
__global__ void swz_kernel(const float* __restrict__ qw, const float* __restrict__ kvw,
                           const float* __restrict__ pw,
                           _Float16* __restrict__ wqkv, _Float16* __restrict__ wproj) {
  int e = blockIdx.x * 256 + threadIdx.x;
  if (e < 196608) {                       // QKV: combined [256 in][768 out]
    int frag = e >> 9, w = e & 511, lane = w >> 4, p = w & 15;
    int nt = frag >> 3, kt = frag & 7;
    int col = nt * 16 + (lane & 15);
    int K   = kt * 32 + ((lane >> 4) << 4) + p;
    float v = (col < 256) ? qw[K * 256 + col] : kvw[K * 512 + (col - 256)];
    wqkv[e] = (_Float16)v;
  } else {
    int e2 = e - 196608;
    if (e2 < 65536) {                     // proj: [256 in][256 out]
      int frag = e2 >> 9, w = e2 & 511, lane = w >> 4, p = w & 15;
      int nt = frag >> 3, kt = frag & 7;
      int col = nt * 16 + (lane & 15);
      int K   = kt * 32 + ((lane >> 4) << 4) + p;
      wproj[e2] = (_Float16)pw[K * 256 + col];
    }
  }
}

// ---------------- rel-pos bias MLP + scatter into [8][64][64] ----------------
__device__ __forceinline__ void mlp_layer(const float* in, float* out,
                                          const float* g, const float* b,
                                          const float* w, const float* bias, int nout) {
  float m = 0.f;
#pragma unroll
  for (int i = 0; i < 16; ++i) m += in[i];
  m *= (1.f / 16.f);
  float v = 0.f;
#pragma unroll
  for (int i = 0; i < 16; ++i) { float d = in[i] - m; v += d * d; }
  v *= (1.f / 16.f);
  float is = rsqrtf(v + 1e-5f);
  float t[16];
#pragma unroll
  for (int i = 0; i < 16; ++i) {
    float y = (in[i] - m) * is * g[i] + b[i];
    t[i] = fmaxf(y, 0.f);
  }
  for (int j = 0; j < nout; ++j) {
    float a = bias[j];
#pragma unroll
    for (int i = 0; i < 16; ++i) a += t[i] * w[i * nout + j];
    out[j] = a;
  }
}

__global__ void pos_kernel(const float* __restrict__ ppw, const float* __restrict__ ppb,
                           const float* __restrict__ g1, const float* __restrict__ b1,
                           const float* __restrict__ w1, const float* __restrict__ bb1,
                           const float* __restrict__ g2, const float* __restrict__ b2,
                           const float* __restrict__ w2, const float* __restrict__ bb2,
                           const float* __restrict__ g3, const float* __restrict__ b3,
                           const float* __restrict__ w3, const float* __restrict__ bb3,
                           float* __restrict__ biasp) {
  __shared__ float posb[169 * 8];
  int t = threadIdx.x;
  if (t < 169) {
    float u[16], y[16];
    float ca = (float)(t / 13) - 6.f;
    float cb = (float)(t % 13) - 6.f;
#pragma unroll
    for (int i = 0; i < 16; ++i) u[i] = ca * ppw[i] + cb * ppw[16 + i] + ppb[i];
    mlp_layer(u, y, g1, b1, w1, bb1, 16);
    mlp_layer(y, u, g2, b2, w2, bb2, 16);
    mlp_layer(u, y, g3, b3, w3, bb3, 8);
    for (int j = 0; j < 8; ++j) posb[t * 8 + j] = y[j];
  }
  __syncthreads();
  for (int idx = t; idx < 32768; idx += 256) {
    int hh = idx >> 12, rem = idx & 4095, i = rem >> 6, j = rem & 63;
    float v = 0.f;
    if (i < 49 && j < 49) {
      int di = i / 7 - j / 7 + 6;
      int dj = i % 7 - j % 7 + 6;
      v = posb[(di * 13 + dj) * 8 + hh];
    }
    biasp[idx] = v;
  }
}

// ---------------- main attention kernel: 1 block = 1 window, 8 waves ----------------
// LDS (dynamic, 128 KB as halfs):
//   [0,     16384): X (x as f16, 64x256)  -> reused as S (8 heads x 64x64, spans [0, 32768)) after QKV
//   [16384, 32768): Q (64x256, row-major) -> reused as O (attn@v result, 64x256) after scores
//   [32768, 49152): K (64x256, row-major)
//   [49152, 65536): VT (8 heads x [32 d][64 m])
__global__ void __launch_bounds__(256, 1)
attn_kernel(const float* __restrict__ x,
            const _Float16* __restrict__ wqkv,
            const _Float16* __restrict__ wproj,
            const float* __restrict__ biasp,
            const float* __restrict__ q_b,
            const float* __restrict__ kv_b,
            const float* __restrict__ proj_b,
            float* __restrict__ outg) {
  extern __shared__ _Float16 smem[];
  _Float16* X  = smem;
  _Float16* Q  = smem + 16384;
  _Float16* K  = smem + 32768;
  _Float16* VT = smem + 49152;
  _Float16* S  = smem;           // alias X (dead after QKV)
  _Float16* O  = smem + 16384;   // alias Q (per-wave disjoint columns)

  const int lane = threadIdx.x & 31;
  const int wv   = threadIdx.x >> 5;
  const int hi   = lane >> 4;
  const int ln   = lane & 15;

  // ---- stage x -> LDS f16, zero-pad rows 49..63 ----
  const float* xg = x + (size_t)blockIdx.x * 12544;
  for (int i = threadIdx.x; i < 3136; i += 256) {
    f4v v = *(const f4v*)(xg + i * 4);
    h4v hv;
    hv[0] = (_Float16)v[0]; hv[1] = (_Float16)v[1];
    hv[2] = (_Float16)v[2]; hv[3] = (_Float16)v[3];
    *(h4v*)(X + i * 4) = hv;
  }
  for (int i = 12544 + threadIdx.x; i < 16384; i += 256) X[i] = (_Float16)0.f;
  __syncthreads();

  // ---- QKV GEMM: [64,256] x [256,768], wave w owns cols [96w, 96w+96) ----
  // N-chunked: 2 chunks of 3 ntiles; acc spans all 4 mtiles -> weights stream once.
#pragma unroll 1
  for (int c = 0; c < 2; ++c) {
    f8v acc[4][3];
#pragma unroll
    for (int mt = 0; mt < 4; ++mt)
#pragma unroll
      for (int n2 = 0; n2 < 3; ++n2) acc[mt][n2] = (f8v){};
#pragma unroll 1
    for (int kt = 0; kt < 8; ++kt) {
      h16v a[4];
#pragma unroll
      for (int mt = 0; mt < 4; ++mt) a[mt] = ldsA(X, mt * 16, 256, kt * 32, lane);
#pragma unroll
      for (int n2 = 0; n2 < 3; ++n2) {
        int nt = wv * 6 + c * 3 + n2;
        h16v b = ldB32(wqkv + (nt * 8 + kt) * 512 + lane * 16);
#pragma unroll
        for (int mt = 0; mt < 4; ++mt) acc[mt][n2] = wmma16(a[mt], b, acc[mt][n2]);
      }
    }
    // epilogue: bias + store q/k row-major, v transposed
#pragma unroll
    for (int n2 = 0; n2 < 3; ++n2) {
      int nt = wv * 6 + c * 3 + n2, cb = nt * 16;
      int col = cb + ln;
      float bias = (cb < 256) ? q_b[col] : kv_b[col - 256];
#pragma unroll
      for (int mt = 0; mt < 4; ++mt) {
        f8v av = acc[mt][n2];
#pragma unroll
        for (int r = 0; r < 8; ++r) av[r] += bias;
        int m0 = mt * 16 + hi * 8;
        if (cb < 512) {                     // q,k row-major (scattered b16 stores)
          _Float16* dst = (cb < 256) ? Q : K;
          int cc = (cb < 256) ? col : (col - 256);
#pragma unroll
          for (int r = 0; r < 8; ++r) dst[(m0 + r) * 256 + cc] = (_Float16)av[r];
        } else {                            // v transposed [d][m]: one packed b128 store
          int f = cb - 512;
          int hh = f >> 5, d = (f & 31) + ln;
          h8v pk;
#pragma unroll
          for (int r = 0; r < 8; ++r) pk[r] = (_Float16)av[r];
          *(h8v*)(VT + hh * 2048 + d * 64 + m0) = pk;
        }
      }
    }
  }
  __syncthreads();

  // ---- scores + softmax: wave w = head w ----
  const int h = wv;
  const float scale = 0.17677669529663687f;  // 32^-0.5
  const float* bp = biasp + h * 4096;
#pragma unroll 1
  for (int mt = 0; mt < 4; ++mt) {
    h16v aq = ldsA(Q, mt * 16, 256, h * 32, lane);
    float Sv[4][8];
#pragma unroll
    for (int nt = 0; nt < 4; ++nt) {
      h16v bk = ldB32(K + (nt * 16 + ln) * 256 + h * 32 + hi * 16);  // B = k^T
      f8v acc = {};
      acc = wmma16(aq, bk, acc);
      int colb = nt * 16 + ln;
      if (colb < 49) {
#pragma unroll
        for (int r = 0; r < 8; ++r)
          Sv[nt][r] = acc[r] * scale + bp[(mt * 16 + hi * 8 + r) * 64 + colb];
      } else {
#pragma unroll
        for (int r = 0; r < 8; ++r) Sv[nt][r] = -1e30f;
      }
    }
    // softmax over key dim: 4 regs/lane x 16 lanes within each half-wave
#pragma unroll
    for (int r = 0; r < 8; ++r) {
      float mx = fmaxf(fmaxf(Sv[0][r], Sv[1][r]), fmaxf(Sv[2][r], Sv[3][r]));
      mx = fmaxf(mx, __shfl_xor(mx, 1));
      mx = fmaxf(mx, __shfl_xor(mx, 2));
      mx = fmaxf(mx, __shfl_xor(mx, 4));
      mx = fmaxf(mx, __shfl_xor(mx, 8));
      float s0 = 0.f;
#pragma unroll
      for (int nt = 0; nt < 4; ++nt) {
        float e = __expf(Sv[nt][r] - mx);
        Sv[nt][r] = e; s0 += e;
      }
      s0 += __shfl_xor(s0, 1);
      s0 += __shfl_xor(s0, 2);
      s0 += __shfl_xor(s0, 4);
      s0 += __shfl_xor(s0, 8);
      float inv = 1.f / s0;
#pragma unroll
      for (int nt = 0; nt < 4; ++nt) Sv[nt][r] *= inv;
    }
    // store probabilities f16 row-major [64 x 64] for this head
#pragma unroll
    for (int nt = 0; nt < 4; ++nt)
#pragma unroll
      for (int r = 0; r < 8; ++r)
        S[h * 4096 + (mt * 16 + hi * 8 + r) * 64 + nt * 16 + ln] = (_Float16)Sv[nt][r];
  }

  // ---- attn @ v : [64,64] x [64,32] per head ----
#pragma unroll 1
  for (int mt = 0; mt < 4; ++mt) {
#pragma unroll
    for (int dt = 0; dt < 2; ++dt) {
      f8v acc = {};
#pragma unroll
      for (int kt = 0; kt < 2; ++kt) {
        h16v as = ldsA(S + h * 4096, mt * 16, 64, kt * 32, lane);
        h16v bv = ldB32(VT + h * 2048 + (dt * 16 + ln) * 64 + kt * 32 + hi * 16);
        acc = wmma16(as, bv, acc);
      }
      int c = h * 32 + dt * 16 + ln, m0 = mt * 16 + hi * 8;
#pragma unroll
      for (int r = 0; r < 8; ++r) O[(m0 + r) * 256 + c] = (_Float16)acc[r];
    }
  }
  __syncthreads();

  // ---- output projection: [64,256] x [256,256], wave w owns cols [32w, 32w+32) ----
  // Single N-chunk, acc spans all 4 mtiles -> proj weights stream once per block.
  float* og = outg + (size_t)blockIdx.x * 12544;
  {
    f8v pacc[4][2];
#pragma unroll
    for (int mt = 0; mt < 4; ++mt)
#pragma unroll
      for (int n2 = 0; n2 < 2; ++n2) pacc[mt][n2] = (f8v){};
#pragma unroll 1
    for (int kt = 0; kt < 8; ++kt) {
      h16v a[4];
#pragma unroll
      for (int mt = 0; mt < 4; ++mt) a[mt] = ldsA(O, mt * 16, 256, kt * 32, lane);
#pragma unroll
      for (int n2 = 0; n2 < 2; ++n2) {
        int nt = wv * 2 + n2;
        h16v b = ldB32(wproj + (nt * 8 + kt) * 512 + lane * 16);
#pragma unroll
        for (int mt = 0; mt < 4; ++mt) pacc[mt][n2] = wmma16(a[mt], b, pacc[mt][n2]);
      }
    }
#pragma unroll
    for (int n2 = 0; n2 < 2; ++n2) {
      int cb = (wv * 2 + n2) * 16;
      float pb = proj_b[cb + ln];
#pragma unroll
      for (int mt = 0; mt < 4; ++mt) {
        int m0 = mt * 16 + hi * 8;
#pragma unroll
        for (int r = 0; r < 8; ++r) {
          int m = m0 + r;
          if (m < 49) og[m * 256 + cb + ln] = pacc[mt][n2][r] + pb;
        }
      }
    }
  }
}

extern "C" void kernel_launch(void* const* d_in, const int* in_sizes, int n_in,
                              void* d_out, int out_size, void* d_ws, size_t ws_size,
                              hipStream_t stream) {
  const float* x      = (const float*)d_in[0];
  const float* q_w    = (const float*)d_in[1];
  const float* q_b    = (const float*)d_in[2];
  const float* kv_w   = (const float*)d_in[3];
  const float* kv_b   = (const float*)d_in[4];
  const float* proj_w = (const float*)d_in[5];
  const float* proj_b = (const float*)d_in[6];
  const float* ppw    = (const float*)d_in[7];
  const float* ppb    = (const float*)d_in[8];
  const float* ln1_g  = (const float*)d_in[9];
  const float* ln1_b  = (const float*)d_in[10];
  const float* w1     = (const float*)d_in[11];
  const float* b1     = (const float*)d_in[12];
  const float* ln2_g  = (const float*)d_in[13];
  const float* ln2_b  = (const float*)d_in[14];
  const float* w2     = (const float*)d_in[15];
  const float* b2     = (const float*)d_in[16];
  const float* ln3_g  = (const float*)d_in[17];
  const float* ln3_b  = (const float*)d_in[18];
  const float* w3     = (const float*)d_in[19];
  const float* b3     = (const float*)d_in[20];

  _Float16* wqkv  = (_Float16*)d_ws;
  _Float16* wproj = (_Float16*)((char*)d_ws + 393216);
  float*    biasp = (float*)((char*)d_ws + 524288);

  swz_kernel<<<1024, 256, 0, stream>>>(q_w, kv_w, proj_w, wqkv, wproj);
  pos_kernel<<<1, 256, 0, stream>>>(ppw, ppb, ln1_g, ln1_b, w1, b1,
                                    ln2_g, ln2_b, w2, b2, ln3_g, ln3_b, w3, b3, biasp);
  attn_kernel<<<4096, 256, 131072, stream>>>(x, wqkv, wproj, biasp,
                                             q_b, kv_b, proj_b, (float*)d_out);
}
